// FaceDetector_45603962749404
// MI455X (gfx1250) — matrix operand order repaired and checked by reading, not compiled, not run on hardware
//
#include <hip/hip_runtime.h>
#include <hip/hip_fp16.h>

typedef __attribute__((ext_vector_type(16))) _Float16 v16h;
typedef __attribute__((ext_vector_type(8)))  _Float16 v8h;
typedef __attribute__((ext_vector_type(8)))  float    v8f;
typedef _Float16 h16;

#define NB 8
#define NANCH 23500
#define TOPK 5000
#define KEEPK 750
#define SORTN 32768

// ---------------- pack f32 pointwise weights -> f16 64x64 zero-padded,
// row-major [m][k] so a WMMA A-fragment is two contiguous 16B reads.
__global__ void k_pack_w(const float* __restrict__ w, h16* __restrict__ wp,
                         int Cout, int Cin) {
  int t = blockIdx.x * blockDim.x + threadIdx.x;
  if (t >= 64 * 64) return;
  int m = t >> 6, k = t & 63;
  float v = (m < Cout && k < Cin) ? w[m * Cin + k] : 0.f;
  wp[t] = (h16)v;
}

// ---------------- fused 1x1 conv (GEMM) via v_wmma_f32_16x16x32_f16.
// Block = 128 threads = 4 waves. Wave w owns M-tile w (16 out channels).
// Block tile: 64(M) x 32(N pixels). Weights (64x64 f16, 8KB) are staged
// into LDS once per block with global_load_async_to_lds_b128 (ASYNCcnt);
// each K-step stages a 32(K)x32(N) B tile into LDS in the wave32
// B-matrix lane layout so a lane consumes one contiguous v16h (32B).
__global__ void __launch_bounds__(128) k_pw_gemm(
    const h16* __restrict__ X, const h16* __restrict__ Wp,
    const float* __restrict__ bias, h16* __restrict__ Y,
    int Cin, int Cout, int HW, int Ksteps) {
  __shared__ __align__(16) h16 ldsW[64 * 64];
  __shared__ __align__(32) h16 ldsB[1024];  // two 32x16 sub-tiles, lane-major
  const int b     = blockIdx.y;
  const int p0    = blockIdx.x * 32;
  const int tid   = threadIdx.x;
  const int lane  = tid & 31;
  const int n     = lane & 15;
  const int khalf = lane >> 4;
  const int mbase = (tid >> 5) * 16;

  const h16* Xb = X + (size_t)b * Cin * HW;
  h16*       Yb = Y + (size_t)b * Cout * HW;

  if (p0 + 32 < HW) __builtin_prefetch(Xb + p0 + 32, 0, 1);  // global_prefetch_b8

  // ---- async-stage packed weights into LDS (gfx1250 async copy path)
#if defined(__gfx1250__)
  {
    unsigned lbase = (unsigned)(uintptr_t)&ldsW[0];
#pragma unroll
    for (int it = 0; it < 4; ++it) {
      unsigned boff = (unsigned)(tid + it * 128) * 16u;
      unsigned long long ga = (unsigned long long)(uintptr_t)((const char*)Wp + boff);
      unsigned la = lbase + boff;
      asm volatile("global_load_async_to_lds_b128 %0, %1, off"
                   :: "v"(la), "v"(ga) : "memory");
    }
    asm volatile("s_wait_asynccnt 0" ::: "memory");
  }
#else
  for (int it = 0; it < 4; ++it) {
    int idx = tid + it * 128;
    *(v8h*)(ldsW + idx * 8) = *(const v8h*)(Wp + idx * 8);
  }
#endif
  __syncthreads();

  v8f c0, c1;
#pragma unroll
  for (int r = 0; r < 8; ++r) {
    int m = mbase + r + 8 * khalf;
    float bv = (m < Cout) ? bias[m] : 0.f;
    c0[r] = bv;
    c1[r] = bv;
  }

  const bool fast = ((HW & 7) == 0) && (p0 + 32 <= HW);

  for (int s = 0; s < Ksteps; ++s) {
    __syncthreads();
    {  // stage B: element (k,nn) -> ldsB[(nn/16)*512 + (nn%16 + (k/16)*16)*16 + k%16]
      int kl = tid >> 2;            // local k 0..31
      int n0 = (tid & 3) * 8;       // 8 consecutive pixels
      int kg = s * 32 + kl;
      int e  = kl & 15;
      int lh = (kl >> 4) << 4;
      if (fast && kg < Cin) {
        v8h xv = *(const v8h*)(Xb + (size_t)kg * HW + p0 + n0);
#pragma unroll
        for (int u = 0; u < 8; ++u) {
          int nn = n0 + u;
          ldsB[((nn >> 4) << 9) + (lh + (nn & 15)) * 16 + e] = xv[u];
        }
      } else {
#pragma unroll
        for (int u = 0; u < 8; ++u) {
          int nn = n0 + u;
          int p  = p0 + nn;
          h16 v = (h16)0.f;
          if (kg < Cin && p < HW) v = Xb[(size_t)kg * HW + p];
          ldsB[((nn >> 4) << 9) + (lh + (nn & 15)) * 16 + e] = v;
        }
      }
    }
    __syncthreads();

    // A fragment from LDS: lane = row m; VGPR0-7 <- K[k0..k0+15] (k0 = 16*khalf)
    v16h a;
    {
      const h16* wr = ldsW + (size_t)(mbase + n) * 64 + s * 32 + khalf * 16;
      v8h lo = *(const v8h*)(wr);
      v8h hi = *(const v8h*)(wr + 8);
#pragma unroll
      for (int e2 = 0; e2 < 8; ++e2) { a[e2] = lo[e2]; a[8 + e2] = hi[e2]; }
    }
    v16h bm0 = *(const v16h*)(ldsB + lane * 16);
    v16h bm1 = *(const v16h*)(ldsB + 512 + lane * 16);
    c0 = __builtin_amdgcn_wmma_f32_16x16x32_f16(false, a, false, bm0,
                                                (short)0, c0, false, false);
    c1 = __builtin_amdgcn_wmma_f32_16x16x32_f16(false, a, false, bm1,
                                                (short)0, c1, false, false);
  }

#pragma unroll
  for (int r = 0; r < 8; ++r) {
    int m = mbase + r + 8 * khalf;
    if (m < Cout) {
      int pA = p0 + n;
      int pB = p0 + 16 + n;
      if (pA < HW) Yb[(size_t)m * HW + pA] = (h16)c0[r];
      if (pB < HW) Yb[(size_t)m * HW + pB] = (h16)c1[r];
    }
  }
}

// ---------------- depthwise 3x3 (pad 1) + bias + optional BN+ReLU
__global__ void k_dw(const h16* __restrict__ X, h16* __restrict__ Y,
                     const float* __restrict__ w2, const float* __restrict__ b2,
                     const float* __restrict__ bng, const float* __restrict__ bnb,
                     const float* __restrict__ bnm, const float* __restrict__ bnv,
                     int C, int H, int W, int has_bn) {
  size_t t = (size_t)blockIdx.x * blockDim.x + threadIdx.x;
  size_t total = (size_t)NB * C * H * W;
  if (t >= total) return;
  int HW = H * W;
  int x = (int)(t % W);
  int y = (int)((t / W) % H);
  int cch = (int)((t / HW) % C);
  int b = (int)(t / ((size_t)C * HW));
  const h16*   Xc = X + ((size_t)b * C + cch) * HW;
  const float* wc = w2 + cch * 9;
  float acc = b2[cch];
#pragma unroll
  for (int ky = 0; ky < 3; ++ky) {
    int iy = y + ky - 1;
    if (iy < 0 || iy >= H) continue;
#pragma unroll
    for (int kx = 0; kx < 3; ++kx) {
      int ix = x + kx - 1;
      if (ix < 0 || ix >= W) continue;
      acc += wc[ky * 3 + kx] * (float)Xc[iy * W + ix];
    }
  }
  if (has_bn) {
    float s = bng[cch] * rsqrtf(bnv[cch] + 1e-5f);
    acc = (acc - bnm[cch]) * s + bnb[cch];
    acc = fmaxf(acc, 0.f);
  }
  Y[((size_t)b * C + cch) * HW + (size_t)y * W + x] = (h16)acc;
}

// ---------------- first conv: 3->16, 3x3 s2 p1, + BN + ReLU, f32 image -> f16
__global__ void k_conv0(const float* __restrict__ img, h16* __restrict__ Y,
                        const float* __restrict__ w, const float* __restrict__ bias,
                        const float* __restrict__ g, const float* __restrict__ bb,
                        const float* __restrict__ m, const float* __restrict__ v) {
  size_t t = (size_t)blockIdx.x * blockDim.x + threadIdx.x;
  if (t >= (size_t)NB * 16 * 320 * 320) return;
  int x = (int)(t % 320);
  int y = (int)((t / 320) % 320);
  int o = (int)((t / (320 * 320)) % 16);
  int b = (int)(t / (16 * 320 * 320));
  const float* ib = img + (size_t)b * 3 * 640 * 640;
  float acc = bias[o];
#pragma unroll
  for (int ci = 0; ci < 3; ++ci)
    for (int ky = 0; ky < 3; ++ky) {
      int iy = 2 * y - 1 + ky;
      if (iy < 0 || iy >= 640) continue;
      for (int kx = 0; kx < 3; ++kx) {
        int ix = 2 * x - 1 + kx;
        if (ix < 0 || ix >= 640) continue;
        acc += w[((o * 3 + ci) * 3 + ky) * 3 + kx] * ib[((size_t)ci * 640 + iy) * 640 + ix];
      }
    }
  float s = g[o] * rsqrtf(v[o] + 1e-5f);
  acc = fmaxf((acc - m[o]) * s + bb[o], 0.f);
  Y[t] = (h16)acc;
}

// ---------------- 2x2 max pool
__global__ void k_pool(const h16* __restrict__ X, h16* __restrict__ Y,
                       int C, int Ho, int Wo) {
  size_t t = (size_t)blockIdx.x * blockDim.x + threadIdx.x;
  size_t total = (size_t)NB * C * Ho * Wo;
  if (t >= total) return;
  int x = (int)(t % Wo);
  int y = (int)((t / Wo) % Ho);
  int cch = (int)((t / ((size_t)Ho * Wo)) % C);
  int b = (int)(t / ((size_t)C * Ho * Wo));
  int Wi = Wo * 2;
  const h16* Xc = X + ((size_t)b * C + cch) * (4 * (size_t)Ho * Wo);
  float a0 = (float)Xc[(size_t)(2 * y) * Wi + 2 * x];
  float a1 = (float)Xc[(size_t)(2 * y) * Wi + 2 * x + 1];
  float a2 = (float)Xc[(size_t)(2 * y + 1) * Wi + 2 * x];
  float a3 = (float)Xc[(size_t)(2 * y + 1) * Wi + 2 * x + 1];
  Y[t] = (h16)fmaxf(fmaxf(a0, a1), fmaxf(a2, a3));
}

// ---------------- gather head outputs, softmax conf, decode with priors
__global__ void k_decode(const h16* __restrict__ h0, const h16* __restrict__ h1,
                         const h16* __restrict__ h2, const h16* __restrict__ h3,
                         float* __restrict__ boxes, float* __restrict__ scores) {
  int t = blockIdx.x * blockDim.x + threadIdx.x;
  if (t >= NB * NANCH) return;
  int b = t / NANCH, a = t % NANCH;
  const h16* hp; int fw, nA, base, step, lvl;
  if (a < 19200)      { hp = h0; fw = 80; nA = 3; base = 0;     step = 8;  lvl = 0; }
  else if (a < 22400) { hp = h1; fw = 40; nA = 2; base = 19200; step = 16; lvl = 1; }
  else if (a < 23200) { hp = h2; fw = 20; nA = 2; base = 22400; step = 32; lvl = 2; }
  else                { hp = h3; fw = 10; nA = 3; base = 23200; step = 64; lvl = 3; }
  int fhw = fw * fw;
  int Cl = nA * 17;
  int aa = a - base;
  int cell = aa / nA;
  int j = aa - cell * nA;
  int px = cell % fw, py = cell / fw;
  const h16* hb = hp + (size_t)b * Cl * fhw;
  float f[17];
#pragma unroll
  for (int q = 0; q < 17; ++q) f[q] = (float)hb[(size_t)(j * 17 + q) * fhw + cell];

  float ms;
  if (lvl == 0)      ms = (j == 0) ? 10.f : (j == 1) ? 16.f : 24.f;
  else if (lvl == 1) ms = (j == 0) ? 32.f : 48.f;
  else if (lvl == 2) ms = (j == 0) ? 64.f : 96.f;
  else               ms = (j == 0) ? 128.f : (j == 1) ? 192.f : 256.f;

  const float inv = 1.f / 640.f;
  float cx = (px + 0.5f) * step * inv;
  float cy = (py + 0.5f) * step * inv;
  float sw = ms * inv, sh = ms * inv;

  float qx = cx + f[0] * 0.1f * sw;
  float qy = cy + f[1] * 0.1f * sh;
  float ww = sw * expf(f[2] * 0.2f);
  float hh = sh * expf(f[3] * 0.2f);
  float* bo = boxes + ((size_t)b * NANCH + a) * 14;
  float x1 = (qx - ww * 0.5f) * 640.f, y1 = (qy - hh * 0.5f) * 640.f;
  bo[0] = x1; bo[1] = y1; bo[2] = x1 + ww * 640.f; bo[3] = y1 + hh * 640.f;
#pragma unroll
  for (int l = 0; l < 5; ++l) {
    bo[4 + 2 * l] = (cx + f[4 + 2 * l] * 0.1f * sw) * 640.f;
    bo[5 + 2 * l] = (cy + f[5 + 2 * l] * 0.1f * sh) * 640.f;
  }
  float p1 = 1.f / (1.f + expf(f[14] - f[15]));  // softmax class-1 prob
  float ic = fminf(fmaxf(f[16], 0.f), 1.f);
  scores[(size_t)b * NANCH + a] = sqrtf(fmaxf(p1 * ic, 1e-12f));
}

// ---------------- sort keys: (orderable -score / 2.0 sentinel) << 32 | index
__global__ void k_sort_init(const float* __restrict__ scores,
                            unsigned long long* __restrict__ sk) {
  int t = blockIdx.x * blockDim.x + threadIdx.x;
  if (t >= NB * SORTN) return;
  int b = t >> 15, i = t & (SORTN - 1);
  float key;
  if (i < NANCH) {
    float s = scores[(size_t)b * NANCH + i];
    key = (s > 0.3f) ? -s : 2.0f;
  } else key = 3.0f;
  unsigned u = __float_as_uint(key);
  u = (u >> 31) ? ~u : (u | 0x80000000u);
  sk[t] = ((unsigned long long)u << 32) | (unsigned)i;
}

__global__ void k_bitonic(unsigned long long* __restrict__ sk, int k, int j) {
  int t = blockIdx.x * blockDim.x + threadIdx.x;
  if (t >= NB * SORTN) return;
  int b = t >> 15, i = t & (SORTN - 1);
  int ix = i ^ j;
  if (ix <= i) return;
  unsigned long long* base = sk + ((size_t)b << 15);
  unsigned long long A = base[i], Bv = base[ix];
  bool up = ((i & k) == 0);
  if ((A > Bv) == up) { base[i] = Bv; base[ix] = A; }
}

// ---------------- NMS over top-5000 + stable select of kept -> 750 dets
__global__ void __launch_bounds__(1024) k_nms(
    const unsigned long long* __restrict__ sk, const float* __restrict__ boxes,
    const float* __restrict__ scores, float* __restrict__ out) {
  __shared__ int keep[TOPK];
  __shared__ float sb[5];   // x1,y1,x2,y2,area of candidate i
  __shared__ int scur;
  int b = blockIdx.x, tid = threadIdx.x;
  const unsigned long long* skb = sk + ((size_t)b << 15);

  float x1[5], y1[5], x2[5], y2[5], ar[5];
  int vld[5];
#pragma unroll
  for (int q = 0; q < 5; ++q) {
    int i = tid + q * 1024;
    if (i < TOPK) {
      int id = (int)(skb[i] & 0xFFFFFFFFull);
      const float* bo = boxes + ((size_t)b * NANCH + id) * 14;
      x1[q] = bo[0]; y1[q] = bo[1]; x2[q] = bo[2]; y2[q] = bo[3];
      ar[q] = (x2[q] - x1[q]) * (y2[q] - y1[q]);
      vld[q] = scores[(size_t)b * NANCH + id] > 0.3f;
      keep[i] = 1;
    } else { vld[q] = 0; x1[q]=y1[q]=x2[q]=y2[q]=ar[q]=0.f; }
  }
  __syncthreads();

  for (int i = 0; i < TOPK; ++i) {
    if (tid == 0) {
      int id = (int)(skb[i] & 0xFFFFFFFFull);
      const float* bo = boxes + ((size_t)b * NANCH + id) * 14;
      sb[0] = bo[0]; sb[1] = bo[1]; sb[2] = bo[2]; sb[3] = bo[3];
      sb[4] = (bo[2] - bo[0]) * (bo[3] - bo[1]);
      scur = keep[i] && (scores[(size_t)b * NANCH + id] > 0.3f);
    }
    __syncthreads();
    if (scur) {
#pragma unroll
      for (int q = 0; q < 5; ++q) {
        int jj = tid + q * 1024;
        if (jj > i && jj < TOPK && keep[jj]) {
          float iw = fmaxf(fminf(sb[2], x2[q]) - fmaxf(sb[0], x1[q]), 0.f);
          float ih = fmaxf(fminf(sb[3], y2[q]) - fmaxf(sb[1], y1[q]), 0.f);
          float inter = iw * ih;
          float iou = inter / (sb[4] + ar[q] - inter);
          if (iou > 0.3f) keep[jj] = 0;
        }
      }
    }
    __syncthreads();
  }

  // kept = keep && valid; assign stable output slots (reuse keep[] as pos)
#pragma unroll
  for (int q = 0; q < 5; ++q) {
    int i = tid + q * 1024;
    if (i < TOPK) keep[i] = (keep[i] && vld[q]) ? 1 : 0;
  }
  __syncthreads();
  if (tid == 0) {
    int cnt = 0;
    for (int i = 0; i < TOPK; ++i) {
      if (keep[i]) { keep[i] = (cnt < KEEPK) ? cnt : -1; cnt++; }
      else keep[i] = -1;
    }
  }
  __syncthreads();
  float* ob = out + (size_t)b * KEEPK * 15;
  for (int i = tid; i < KEEPK * 15; i += 1024) ob[i] = 0.f;
  __syncthreads();
#pragma unroll
  for (int q = 0; q < 5; ++q) {
    int i = tid + q * 1024;
    if (i < TOPK) {
      int p = keep[i];
      if (p >= 0) {
        int id = (int)(skb[i] & 0xFFFFFFFFull);
        const float* bo = boxes + ((size_t)b * NANCH + id) * 14;
        for (int f2 = 0; f2 < 14; ++f2) ob[p * 15 + f2] = bo[f2];
        ob[p * 15 + 14] = scores[(size_t)b * NANCH + id];
      }
    }
  }
}

// =====================================================================
extern "C" void kernel_launch(void* const* d_in, const int* in_sizes, int n_in,
                              void* d_out, int out_size, void* d_ws, size_t ws_size,
                              hipStream_t stream) {
  if (n_in < 159) return;  // expect 159 leaves (JAX sorted-key pytree order)
  const float* img = (const float*)d_in[0];
  int ix = 1;
  auto F = [&](int i) { return (const float*)d_in[i]; };

  struct DP { const float *b1, *b2, *bnb, *bng, *bnm, *bnv, *w1, *w2; int bn; };
  auto rd_dp = [&](int bn) {
    DP d{}; d.b1 = F(ix++); d.b2 = F(ix++);
    if (bn) { d.bnb = F(ix++); d.bng = F(ix++); d.bnm = F(ix++); d.bnv = F(ix++); }
    d.w1 = F(ix++); d.w2 = F(ix++); d.bn = bn; return d;
  };
  struct BLK { DP u1, u2; };
  auto rd_blk = [&](int bn2) { BLK bl; bl.u1 = rd_dp(1); bl.u2 = rd_dp(bn2); return bl; };

  // leaf order: image, head(h0..h3), model0(b,bn,u,w), model1..model6
  BLK hB[4];
  for (int k = 0; k < 4; ++k) hB[k] = rd_blk(0);
  const float* c0b = F(ix++);
  const float* c0bnb = F(ix++); const float* c0bng = F(ix++);
  const float* c0bnm = F(ix++); const float* c0bnv = F(ix++);
  DP m0u = rd_dp(1);
  const float* c0w = F(ix++);
  BLK mB[6];
  for (int k = 0; k < 6; ++k) mB[k] = rd_blk(1);

  // ---- workspace carve
  size_t off = 0;
  auto alloc = [&](size_t bytes) {
    off = (off + 255) & ~(size_t)255;
    void* p = (char*)d_ws + off;
    off += bytes;
    return p;
  };
  const size_t BIG = (size_t)NB * 64 * 160 * 160;  // == 8*16*320*320 elements
  h16* bufA = (h16*)alloc(BIG * 2);
  h16* bufB = (h16*)alloc(BIG * 2);
  h16* Wp   = (h16*)alloc(64 * 64 * 2);
  h16* s0 = (h16*)alloc((size_t)NB * 64 * 6400 * 2);
  h16* s1 = (h16*)alloc((size_t)NB * 64 * 1600 * 2);
  h16* s2 = (h16*)alloc((size_t)NB * 64 * 400 * 2);
  h16* s3 = (h16*)alloc((size_t)NB * 64 * 100 * 2);
  h16* ho0 = (h16*)alloc((size_t)NB * 51 * 6400 * 2);
  h16* ho1 = (h16*)alloc((size_t)NB * 34 * 1600 * 2);
  h16* ho2 = (h16*)alloc((size_t)NB * 34 * 400 * 2);
  h16* ho3 = (h16*)alloc((size_t)NB * 51 * 100 * 2);
  float* boxes  = (float*)alloc((size_t)NB * NANCH * 14 * 4);
  float* scores = (float*)alloc((size_t)NB * NANCH * 4);
  unsigned long long* sk = (unsigned long long*)alloc((size_t)NB * SORTN * 8);
  if (off > ws_size) return;

  auto run_pw = [&](const h16* X, h16* Y, const float* w1, const float* b1,
                    int Cin, int Cout, int HW) {
    k_pack_w<<<(4096 + 255) / 256, 256, 0, stream>>>(w1, Wp, Cout, Cin);
    dim3 g((HW + 31) / 32, NB);
    k_pw_gemm<<<g, 128, 0, stream>>>(X, Wp, b1, Y, Cin, Cout, HW, (Cin > 32) ? 2 : 1);
  };
  auto run_dw = [&](const h16* X, h16* Y, const DP& d, int C, int H, int W) {
    size_t total = (size_t)NB * C * H * W;
    k_dw<<<(unsigned)((total + 255) / 256), 256, 0, stream>>>(
        X, Y, d.w2, d.b2, d.bng, d.bnb, d.bnm, d.bnv, C, H, W, d.bn);
  };
  auto run_dp2 = [&](const h16* X, h16* T, h16* Y, const DP& d,
                     int Cin, int Cout, int H, int W) {
    run_pw(X, T, d.w1, d.b1, Cin, Cout, H * W);
    run_dw(T, Y, d, Cout, H, W);
  };
  auto run_pool = [&](const h16* X, h16* Y, int C, int Ho, int Wo) {
    size_t total = (size_t)NB * C * Ho * Wo;
    k_pool<<<(unsigned)((total + 255) / 256), 256, 0, stream>>>(X, Y, C, Ho, Wo);
  };

  // ---- backbone
  {
    size_t nthr = (size_t)NB * 16 * 320 * 320;
    k_conv0<<<(unsigned)((nthr + 255) / 256), 256, 0, stream>>>(
        img, bufA, c0w, c0b, c0bng, c0bnb, c0bnm, c0bnv);
  }
  run_dp2(bufA, bufB, bufA, m0u, 16, 16, 320, 320);     // model0.u @320
  run_pool(bufA, bufB, 16, 160, 160);                   // ->160
  run_dp2(bufB, bufA, bufB, mB[0].u1, 16, 16, 160, 160);
  run_dp2(bufB, bufA, bufB, mB[0].u2, 16, 64, 160, 160);
  run_dp2(bufB, bufA, bufB, mB[1].u1, 64, 64, 160, 160);
  run_dp2(bufB, bufA, bufB, mB[1].u2, 64, 64, 160, 160);
  run_pool(bufB, bufA, 64, 80, 80);                     // ->80
  run_dp2(bufA, bufB, bufA, mB[2].u1, 64, 64, 80, 80);
  run_dp2(bufA, bufB, s0,   mB[2].u2, 64, 64, 80, 80);  // src0
  run_pool(s0, bufA, 64, 40, 40);
  run_dp2(bufA, bufB, bufA, mB[3].u1, 64, 64, 40, 40);
  run_dp2(bufA, bufB, s1,   mB[3].u2, 64, 64, 40, 40);  // src1
  run_pool(s1, bufA, 64, 20, 20);
  run_dp2(bufA, bufB, bufA, mB[4].u1, 64, 64, 20, 20);
  run_dp2(bufA, bufB, s2,   mB[4].u2, 64, 64, 20, 20);  // src2
  run_pool(s2, bufA, 64, 10, 10);
  run_dp2(bufA, bufB, bufA, mB[5].u1, 64, 64, 10, 10);
  run_dp2(bufA, bufB, s3,   mB[5].u2, 64, 64, 10, 10);  // src3

  // ---- heads
  const h16* srcs[4] = {s0, s1, s2, s3};
  h16* hos[4] = {ho0, ho1, ho2, ho3};
  const int fws[4] = {80, 40, 20, 10};
  const int cls[4] = {51, 34, 34, 51};
  for (int l = 0; l < 4; ++l) {
    int fdim = fws[l];
    run_dp2(srcs[l], bufA, bufB, hB[l].u1, 64, 64, fdim, fdim);
    run_dp2(bufB, bufA, hos[l], hB[l].u2, 64, cls[l], fdim, fdim);
  }

  // ---- decode + score
  {
    int nthr = NB * NANCH;
    k_decode<<<(nthr + 255) / 256, 256, 0, stream>>>(ho0, ho1, ho2, ho3, boxes, scores);
  }
  // ---- per-batch bitonic sort (global memory, 120 stages)
  {
    int nthr = NB * SORTN;
    k_sort_init<<<(nthr + 255) / 256, 256, 0, stream>>>(scores, sk);
    for (int k = 2; k <= SORTN; k <<= 1)
      for (int j = k >> 1; j > 0; j >>= 1)
        k_bitonic<<<(nthr + 255) / 256, 256, 0, stream>>>(sk, k, j);
  }
  // ---- NMS + select
  k_nms<<<NB, 1024, 0, stream>>>(sk, boxes, scores, (float*)d_out);
  (void)in_sizes; (void)out_size;
}
